// BertSelfAttention_55327768707542
// MI455X (gfx1250) — matrix operand order repaired and checked by reading, not compiled, not run on hardware
//
#include <hip/hip_runtime.h>

// ---------------------------------------------------------------------------
// BERT self-attention with relative_key_query position bias, MI455X (gfx1250)
// All matmuls (QKV proj, q.kT, q.E^T, k.E^T, p.v) run on v_wmma_f32_16x16x32_f16
// with f32 accumulation; flash-style online softmax so scores are never
// materialized. One wave32 per workgroup owns a 16-row q tile.
// ---------------------------------------------------------------------------

#define HIDDEN   1024
#define HEADS    16
#define HEAD_DIM 64
#define MAX_POS  2048
#define BATCH    2
#define SEQ      2048
#define MSEQ     (BATCH * SEQ)   // 4096 rows for the QKV GEMM
#define EPAD     4096            // dist_emb rows padded 4095 -> 4096 (row 4095 = 0)

typedef __attribute__((ext_vector_type(16))) _Float16 v16h;
typedef __attribute__((ext_vector_type(8)))  _Float16 v8h;
typedef __attribute__((ext_vector_type(8)))  float    v8f;

static __device__ __forceinline__ v8f wmma_f16(v16h a, v16h b, v8f c) {
  // D = A(16x32) * B(32x16) + C, f32 accumulate
  return __builtin_amdgcn_wmma_f32_16x16x32_f16(false, a, false, b, (short)0, c,
                                                false, false);
}

// A-operand fragment: 16(M) x 32(K) from a row-major matrix, ld in elements.
// lane<16: row=lane,    K 0..15 contiguous
// lane>=16: row=lane-16, K 16..31 contiguous
static __device__ __forceinline__ v16h load_fragA(const _Float16* tile, int ld, int lane) {
  int row  = lane & 15;
  int koff = (lane >> 4) << 4;           // 0 or 16
  const _Float16* p = tile + (size_t)row * ld + koff;
  return *(const v16h*)(p);              // K koff .. koff+15
}

// B-operand fragment: 32(K) x 16(N); source is row-major N x K so that
// column n of B == source row n.  lane<16: col=lane, K 0..15 contiguous;
// lane>=16: col=lane-16, K 16..31 contiguous.
static __device__ __forceinline__ v16h load_fragB(const _Float16* src, int ld, int lane) {
  int n    = lane & 15;
  int koff = (lane >> 4) << 4;           // 0 or 16
  return *(const v16h*)(src + (size_t)n * ld + koff);
}

// f32 -> f16 converting loaders for the projection GEMM
static __device__ __forceinline__ v16h load_fragA_f32(const float* tile, int ld, int lane) {
  int row  = lane & 15;
  int koff = (lane >> 4) << 4;
  const float* p = tile + (size_t)row * ld + koff;
  union { v16h v; _Float16 e[16]; } u;
#pragma unroll
  for (int i = 0; i < 16; ++i) u.e[i] = (_Float16)p[i];
  return u.v;
}
static __device__ __forceinline__ v16h load_fragB_f32(const float* src, int ld, int lane) {
  int n    = lane & 15;
  int koff = (lane >> 4) << 4;
  const float* p = src + (size_t)n * ld + koff;
  union { v16h v; _Float16 e[16]; } u;
#pragma unroll
  for (int i = 0; i < 16; ++i) u.e[i] = (_Float16)p[i];
  return u.v;
}

// reductions across the 16 lanes of each wave half (D-fragment row groups)
static __device__ __forceinline__ float half_max(float v) {
#pragma unroll
  for (int m = 1; m <= 8; m <<= 1) v = fmaxf(v, __shfl_xor(v, m, 32));
  return v;
}
static __device__ __forceinline__ float half_sum(float v) {
#pragma unroll
  for (int m = 1; m <= 8; m <<= 1) v += __shfl_xor(v, m, 32);
  return v;
}

// ---------------------------------------------------------------------------
// dist_emb f32[4095][64] -> f16[4096][64] with zeroed pad row 4095
__global__ __launch_bounds__(256) void cvt_e_kernel(const float* __restrict__ de,
                                                    _Float16* __restrict__ eh) {
  int i = blockIdx.x * 256 + threadIdx.x;
  if (i < EPAD * HEAD_DIM) {
    int row = i >> 6;
    eh[i] = (row < 2 * MAX_POS - 1) ? (_Float16)de[i] : (_Float16)0.0f;
  }
}

// ---------------------------------------------------------------------------
// QKV projection: out[m,n] = sum_k X[m,k] * W[n,k] + b[n]   (torch Linear)
// blockIdx = (n_tile, m_tile, proj).  Writes f16 Q/K row-major [B*S][1024]
// and V transposed [b][h][d][s] so attention B-operands are contiguous.
__global__ __launch_bounds__(32) void qkv_kernel(
    const float* __restrict__ X,
    const float* __restrict__ Wq, const float* __restrict__ bq,
    const float* __restrict__ Wk, const float* __restrict__ bk,
    const float* __restrict__ Wv, const float* __restrict__ bvp,
    _Float16* __restrict__ qh, _Float16* __restrict__ kh,
    _Float16* __restrict__ vT) {
  const int lane = threadIdx.x & 31;
  const int n0   = blockIdx.x * 16;
  const int m0   = blockIdx.y * 16;
  const int proj = blockIdx.z;
  const float* W    = (proj == 0) ? Wq : (proj == 1) ? Wk : Wv;
  const float* bias = (proj == 0) ? bq : (proj == 1) ? bk : bvp;

  v8f acc = {};
  for (int k = 0; k < HIDDEN; k += 32) {
    v16h a = load_fragA_f32(X + (size_t)m0 * HIDDEN + k, HIDDEN, lane);
    v16h b = load_fragB_f32(W + (size_t)n0 * HIDDEN + k, HIDDEN, lane);
    acc = wmma_f16(a, b, acc);
  }
  const int col = lane & 15;
  const int rb  = (lane >> 4) << 3;   // D rows rb..rb+7 for this lane
  const float bn = bias[n0 + col];
  if (proj < 2) {
    _Float16* dst = (proj == 0) ? qh : kh;
#pragma unroll
    for (int r = 0; r < 8; ++r)
      dst[(size_t)(m0 + rb + r) * HIDDEN + n0 + col] = (_Float16)(acc[r] + bn);
  } else {
    int n = n0 + col;
    int h = n >> 6, d = n & 63;
#pragma unroll
    for (int r = 0; r < 8; ++r) {
      int m = m0 + rb + r;
      int b = m >> 11;              // / SEQ
      int s = m & (SEQ - 1);
      vT[(((size_t)b * HEADS + h) * HEAD_DIM + d) * SEQ + s] = (_Float16)(acc[r] + bn);
    }
  }
}

// ---------------------------------------------------------------------------
// Flash attention with relative_key_query bias.
// blockIdx = (q_tile, b*16+h).  One wave handles 16 q rows over all 2048 k
// in 32-wide k tiles.  Per tile: 4 WMMA (q.kT) + 6 WMMA (G=Q*E^T) +
// 12 WMMA (H=K*E^T) + 4 WMMA (p.v) = 26 WMMAs.
__global__ __launch_bounds__(32) void attn_kernel(
    const _Float16* __restrict__ qh, const _Float16* __restrict__ kh,
    const _Float16* __restrict__ vT, const _Float16* __restrict__ eh,
    float* __restrict__ out) {
  __shared__ __align__(32) float    Gs[16 * 48];   // Q x E_window^T
  __shared__ __align__(32) float    Hs[32 * 48];   // K x E_window^T
  __shared__ __align__(32) _Float16 Pls[16 * 32];  // softmax probs, A-staging

  const int lane = threadIdx.x & 31;
  const int q0   = blockIdx.x * 16;
  const int bh   = blockIdx.y;
  const int b    = bh >> 4;
  const int h    = bh & 15;

  const _Float16* Qb = qh + (size_t)b * SEQ * HIDDEN + h * HEAD_DIM;  // row ld=HIDDEN
  const _Float16* Kb = kh + (size_t)b * SEQ * HIDDEN + h * HEAD_DIM;
  const _Float16* Vt = vT + (size_t)bh * HEAD_DIM * SEQ;              // [d][s] ld=SEQ

  // Q tile as A-operand, both d-halves (d 0..31, 32..63); reused every k tile
  const v16h qa0 = load_fragA(Qb + (size_t)q0 * HIDDEN, HIDDEN, lane);
  const v16h qa1 = load_fragA(Qb + (size_t)q0 * HIDDEN + 32, HIDDEN, lane);

  const int col = lane & 15;
  const int rb  = (lane >> 4) << 3;   // this lane's D rows: rb..rb+7

  float mrow[8], lrow[8];
  v8f o0 = {}, o1 = {}, o2 = {}, o3 = {};
#pragma unroll
  for (int r = 0; r < 8; ++r) { mrow[r] = -3.0e30f; lrow[r] = 0.0f; }

  for (int k0 = 0; k0 < SEQ; k0 += 32) {
    if (k0 + 32 < SEQ)
      __builtin_prefetch(Kb + (size_t)(k0 + 32) * HIDDEN, 0, 3);

    // ---- scores: q . k^T  (two 16-col fragments) ----
    v8f s0 = {}, s1 = {};
    {
      v16h b00 = load_fragB(Kb + (size_t)k0 * HIDDEN, HIDDEN, lane);
      v16h b01 = load_fragB(Kb + (size_t)k0 * HIDDEN + 32, HIDDEN, lane);
      s0 = wmma_f16(qa0, b00, s0);
      s0 = wmma_f16(qa1, b01, s0);
      v16h b10 = load_fragB(Kb + (size_t)(k0 + 16) * HIDDEN, HIDDEN, lane);
      v16h b11 = load_fragB(Kb + (size_t)(k0 + 16) * HIDDEN + 32, HIDDEN, lane);
      s1 = wmma_f16(qa0, b10, s1);
      s1 = wmma_f16(qa1, b11, s1);
    }

    // ---- relative-position window: t(i,j) = 2047 + (q0+i) - (k0+j) ----
    const int t0 = (MAX_POS - 1) + q0 - k0 - 31;   // window base, t' = i-j+31
    v16h e0[3], e1[3];
#pragma unroll
    for (int w = 0; w < 3; ++w) {
      e0[w] = load_fragB(eh + (size_t)(t0 + w * 16) * HEAD_DIM, HEAD_DIM, lane);
      e1[w] = load_fragB(eh + (size_t)(t0 + w * 16) * HEAD_DIM + 32, HEAD_DIM, lane);
    }
    // G = Q_tile x E_window^T  (16 x 48)
#pragma unroll
    for (int w = 0; w < 3; ++w) {
      v8f g = {};
      g = wmma_f16(qa0, e0[w], g);
      g = wmma_f16(qa1, e1[w], g);
#pragma unroll
      for (int r = 0; r < 8; ++r) Gs[(rb + r) * 48 + w * 16 + col] = g[r];
    }
    // H = K_tile x E_window^T  (32 x 48)
#pragma unroll
    for (int rg = 0; rg < 2; ++rg) {
      v16h ka0 = load_fragA(Kb + (size_t)(k0 + rg * 16) * HIDDEN, HIDDEN, lane);
      v16h ka1 = load_fragA(Kb + (size_t)(k0 + rg * 16) * HIDDEN + 32, HIDDEN, lane);
#pragma unroll
      for (int w = 0; w < 3; ++w) {
        v8f hh = {};
        hh = wmma_f16(ka0, e0[w], hh);
        hh = wmma_f16(ka1, e1[w], hh);
#pragma unroll
        for (int r = 0; r < 8; ++r) Hs[(rg * 16 + rb + r) * 48 + w * 16 + col] = hh[r];
      }
    }

    // ---- assemble scores + online softmax (DS ops are in-order per wave) ----
    float sc0[8], sc1[8];
#pragma unroll
    for (int r = 0; r < 8; ++r) {
      int il = rb + r;
      int tp0 = il - col + 31;          // j = col       -> t' in [16,46]
      int tp1 = il - (16 + col) + 31;   // j = 16 + col  -> t' in [0,30]
      sc0[r] = s0[r] * 0.125f + Gs[il * 48 + tp0] + Hs[col * 48 + tp0];
      sc1[r] = s1[r] * 0.125f + Gs[il * 48 + tp1] + Hs[(16 + col) * 48 + tp1];
    }
    float alpha[8];
#pragma unroll
    for (int r = 0; r < 8; ++r) {
      float tmax = half_max(fmaxf(sc0[r], sc1[r]));
      float nm   = fmaxf(mrow[r], tmax);
      alpha[r]   = __expf(mrow[r] - nm);
      mrow[r]    = nm;
      sc0[r]     = __expf(sc0[r] - nm);
      sc1[r]     = __expf(sc1[r] - nm);
      lrow[r]    = lrow[r] * alpha[r] + half_sum(sc0[r] + sc1[r]);
    }
#pragma unroll
    for (int r = 0; r < 8; ++r) {
      o0[r] *= alpha[r]; o1[r] *= alpha[r]; o2[r] *= alpha[r]; o3[r] *= alpha[r];
      Pls[(rb + r) * 32 + col]      = (_Float16)sc0[r];
      Pls[(rb + r) * 32 + 16 + col] = (_Float16)sc1[r];
    }

    // ---- p . v : reload P as A-operand, V^T rows as B-operand ----
    v16h pa = load_fragA(Pls, 32, lane);
    o0 = wmma_f16(pa, load_fragB(Vt + (size_t)(0 * 16) * SEQ + k0, SEQ, lane), o0);
    o1 = wmma_f16(pa, load_fragB(Vt + (size_t)(1 * 16) * SEQ + k0, SEQ, lane), o1);
    o2 = wmma_f16(pa, load_fragB(Vt + (size_t)(2 * 16) * SEQ + k0, SEQ, lane), o2);
    o3 = wmma_f16(pa, load_fragB(Vt + (size_t)(3 * 16) * SEQ + k0, SEQ, lane), o3);
  }

  // ---- epilogue: normalize and store ctx[b, q, h, d] ----
  float* Ob = out + (size_t)b * SEQ * HIDDEN + h * HEAD_DIM;
#pragma unroll
  for (int r = 0; r < 8; ++r) {
    float inv = 1.0f / lrow[r];
    size_t ro = (size_t)(q0 + rb + r) * HIDDEN;
    Ob[ro + 0 * 16 + col] = o0[r] * inv;
    Ob[ro + 1 * 16 + col] = o1[r] * inv;
    Ob[ro + 2 * 16 + col] = o2[r] * inv;
    Ob[ro + 3 * 16 + col] = o3[r] * inv;
  }
}

// ---------------------------------------------------------------------------
extern "C" void kernel_launch(void* const* d_in, const int* in_sizes, int n_in,
                              void* d_out, int out_size, void* d_ws, size_t ws_size,
                              hipStream_t stream) {
  (void)in_sizes; (void)n_in; (void)out_size; (void)ws_size;
  const float* X  = (const float*)d_in[0];
  const float* Wq = (const float*)d_in[1];
  const float* bq = (const float*)d_in[2];
  const float* Wk = (const float*)d_in[3];
  const float* bk = (const float*)d_in[4];
  const float* Wv = (const float*)d_in[5];
  const float* bv = (const float*)d_in[6];
  const float* de = (const float*)d_in[7];

  // workspace: qh | kh | vT | eh  (~24.5 MiB total)
  _Float16* qh = (_Float16*)d_ws;
  _Float16* kh = qh + (size_t)MSEQ * HIDDEN;
  _Float16* vT = kh + (size_t)MSEQ * HIDDEN;
  _Float16* eh = vT + (size_t)MSEQ * HIDDEN;

  cvt_e_kernel<<<dim3((EPAD * HEAD_DIM + 255) / 256), dim3(256), 0, stream>>>(de, eh);
  qkv_kernel<<<dim3(HIDDEN / 16, MSEQ / 16, 3), dim3(32), 0, stream>>>(
      X, Wq, bq, Wk, bk, Wv, bv, qh, kh, vT);
  attn_kernel<<<dim3(SEQ / 16, BATCH * HEADS), dim3(32), 0, stream>>>(
      qh, kh, vT, eh, (float*)d_out);
}
